// FusedThorExpert_15564961481508
// MI455X (gfx1250) — matrix operand order repaired and checked by reading, not compiled, not run on hardware
//
#include <hip/hip_runtime.h>
#include <hip/hip_bf16.h>
#include <math.h>

// Problem dims (fixed by the reference)
#define NE 16     // experts
#define NC 512    // tokens per expert (M)
#define NH 1024   // hidden
#define NI 4096   // intermediate

// LDS row stride in dwords: 16 k-pairs + 4 pad -> keeps 16B alignment for
// b128 fragment loads while still skewing banks between rows.
#define LSTRIDE 20

#if __has_builtin(__builtin_amdgcn_global_load_async_to_lds_b128)
#define HAVE_ASYNC_LDS 1
#else
#define HAVE_ASYNC_LDS 0
#endif

typedef __attribute__((ext_vector_type(16))) __bf16 v16bf;
typedef __attribute__((ext_vector_type(8)))  float  v8f;
typedef __attribute__((ext_vector_type(4)))  int    v4i;

typedef __attribute__((address_space(1))) v4i* gv4i_p;  // global v4i*
typedef __attribute__((address_space(3))) v4i* lv4i_p;  // LDS v4i*

struct alignas(16) U4 { unsigned int a, b, c, d; };
struct alignas(8)  U2 { unsigned int a, b; };
struct alignas(16) F4 { float x, y, z, w; };

union Frag {               // naturally 32B-aligned via v16bf
    unsigned int u[8];
    v16bf        v;
};

// fp32 -> bf16 (round-half-up; scalar, for the epilogue)
__device__ __forceinline__ unsigned int f2bf1(float f) {
    return (__float_as_uint(f) + 0x8000u) >> 16;
}

// Pack two fp32 -> packed bf16 pair {hi,lo}: 2 adds + 1 v_perm_b32.
#if __has_builtin(__builtin_amdgcn_perm)
__device__ __forceinline__ unsigned int packbf(float lo, float hi) {
    unsigned int a = __float_as_uint(lo) + 0x8000u;
    unsigned int b = __float_as_uint(hi) + 0x8000u;
    return __builtin_amdgcn_perm(b, a, 0x07060302u);  // {b[3],b[2],a[3],a[2]}
}
#else
__device__ __forceinline__ unsigned int packbf(float lo, float hi) {
    return f2bf1(lo) | (f2bf1(hi) << 16);
}
#endif

// Stage one 128x32 A tile and one 32x128 B tile into LDS as packed bf16 pairs.
// sA layout: [m][kp], row stride LSTRIDE dwords (kp = k/2, 16 pairs per row)
// sB layout: [n][kp], row stride LSTRIDE dwords (B transposed during staging)
// All global accesses are b128.
template<int KK, int NN, bool A_BF16>
__device__ __forceinline__ void stage_tiles(
    unsigned int* __restrict__ sA, unsigned int* __restrict__ sB,
    const void* __restrict__ A, const float* __restrict__ Bf,
    int tid, size_t aBase, int kt, int n0)
{
    if (A_BF16) {
        // A already packed bf16 pairs: direct 16B copies (async when available)
        const unsigned int* Ap = (const unsigned int*)A + (aBase >> 1) + (kt >> 1);
        #pragma unroll
        for (int i = 0; i < 2; ++i) {
            int u = i * 256 + tid;
            int m = u >> 2, q = u & 3;                       // q: dword-quad in row
            const unsigned int* g = Ap + (size_t)m * (KK / 2) + q * 4;
            unsigned int* l = &sA[m * LSTRIDE + q * 4];
#if HAVE_ASYNC_LDS
            __builtin_amdgcn_global_load_async_to_lds_b128(
                (gv4i_p)g, (lv4i_p)l, 0, 0);
#else
            *(U4*)l = *(const U4*)g;
#endif
        }
    } else {
        const float* Ap = (const float*)A + aBase + kt;
        #pragma unroll
        for (int i = 0; i < 4; ++i) {
            int u = i * 256 + tid;
            int m = u >> 3, kq = u & 7;                      // kq: k-quad in row
            F4 f = *(const F4*)(Ap + (size_t)m * KK + kq * 4);
            U2 d;
            d.a = packbf(f.x, f.y);
            d.b = packbf(f.z, f.w);
            *(U2*)&sA[m * LSTRIDE + kq * 2] = d;
        }
    }
    // B tile: two adjacent K rows loaded as b128 each, packed along K,
    // stored transposed into 4 LDS rows.
    #pragma unroll
    for (int i = 0; i < 2; ++i) {
        int u = i * 256 + tid;
        int nq = u & 31, kp = u >> 5;                        // nq: n-quad
        const float* g = Bf + (size_t)(kt + 2 * kp) * NN + n0 + nq * 4;
        F4 r0 = *(const F4*)g;
        F4 r1 = *(const F4*)(g + NN);
        unsigned int* s = &sB[(nq * 4) * LSTRIDE + kp];
        s[0 * LSTRIDE] = packbf(r0.x, r1.x);
        s[1 * LSTRIDE] = packbf(r0.y, r1.y);
        s[2 * LSTRIDE] = packbf(r0.z, r1.z);
        s[3 * LSTRIDE] = packbf(r0.w, r1.w);
    }
}

// Grouped GEMM: Out[e] = act(A[e] (MxK) * B[e] (KxN) + bias[e])
// GELU_OUT: apply exact-erf GELU and store bf16 (intermediate h), else f32.
template<int MM, int NN, int KK, bool A_BF16, bool GELU_OUT>
__global__ __launch_bounds__(256, 1)
void moe_gemm_kernel(const void* __restrict__ A, const float* __restrict__ B,
                     const float* __restrict__ bias, void* __restrict__ Out)
{
    __shared__ unsigned int sA[2][128 * LSTRIDE];
    __shared__ unsigned int sB[2][128 * LSTRIDE];

    const int e    = blockIdx.z;
    const int m0   = blockIdx.y * 128;
    const int n0   = blockIdx.x * 128;
    const int tid  = threadIdx.x;
    const int lane = tid & 31;
    const int wid  = tid >> 5;
    const int l16  = lane >> 4;    // half-wave select
    const int lm   = lane & 15;
    const int wm   = (wid & 1) * 64;   // wave's M offset in block tile
    const int wn   = (wid >> 1) * 32;  // wave's N offset in block tile

    const float* Bf = B + (size_t)e * KK * NN;
    const size_t aBase = (size_t)e * MM * KK + (size_t)m0 * KK;

    v8f acc[4][2];
    {
        v8f z = {};
        #pragma unroll
        for (int mi = 0; mi < 4; ++mi)
            #pragma unroll
            for (int ni = 0; ni < 2; ++ni)
                acc[mi][ni] = z;
    }

    stage_tiles<KK, NN, A_BF16>(sA[0], sB[0], A, Bf, tid, aBase, 0, n0);
#if HAVE_ASYNC_LDS
    if (A_BF16) asm volatile("s_wait_asynccnt 0" ::: "memory");
#endif
    __syncthreads();

    const int nk = KK >> 5;  // K-steps of 32
    #pragma unroll 1
    for (int kt = 0; kt < nk; ++kt) {
        const int buf = kt & 1;
        if (kt + 1 < nk)
            stage_tiles<KK, NN, A_BF16>(sA[buf ^ 1], sB[buf ^ 1], A, Bf, tid,
                                        aBase, (kt + 1) << 5, n0);

        // Prefetch the B tile two K-steps ahead (weights dominate HBM traffic).
        if (kt + 2 < nk) {
            const int kpref = (kt + 2) << 5;
            const float* pf = Bf + (size_t)(kpref + (tid >> 3)) * NN + n0 + ((tid & 7) << 4);
            __builtin_prefetch(pf, 0, 1);
        }

        Frag a[4], b[2];
        // A fragment (16x32 bf16): lane = row M; half-wave splits K pairs.
        // kp(v) = (v>>2)*8 + l16*4 + (v&3)  -> two aligned 16B runs.
        #pragma unroll
        for (int mi = 0; mi < 4; ++mi) {
            const int m = wm + mi * 16 + lm;
            const unsigned int* base = &sA[buf][m * LSTRIDE + l16 * 4];
            *(U4*)&a[mi].u[0] = *(const U4*)(base);
            *(U4*)&a[mi].u[4] = *(const U4*)(base + 8);
        }
        // B fragment (32x16 bf16): lane = column N; half-wave splits K range.
        // kp(v) = (v>>2)*8 + l16*4 + (v&3)  -> two aligned 16B runs.
        #pragma unroll
        for (int ni = 0; ni < 2; ++ni) {
            const int n = wn + ni * 16 + lm;
            const unsigned int* base = &sB[buf][n * LSTRIDE + l16 * 4];
            *(U4*)&b[ni].u[0] = *(const U4*)(base);
            *(U4*)&b[ni].u[4] = *(const U4*)(base + 8);
        }
        #pragma unroll
        for (int mi = 0; mi < 4; ++mi)
            #pragma unroll
            for (int ni = 0; ni < 2; ++ni)
                acc[mi][ni] = __builtin_amdgcn_wmma_f32_16x16x32_bf16(
                    false, a[mi].v, false, b[ni].v,
                    (short)0, acc[mi][ni], false, false);

#if HAVE_ASYNC_LDS
        if (A_BF16) asm volatile("s_wait_asynccnt 0" ::: "memory");
#endif
        __syncthreads();
    }

    // Epilogue: C/D layout — VGPR r holds (M = r + 8*l16, N = lane%16)
    #pragma unroll
    for (int ni = 0; ni < 2; ++ni) {
        const int ncol = n0 + wn + ni * 16 + lm;
        const float bv = bias[(size_t)e * NN + ncol];
        #pragma unroll
        for (int mi = 0; mi < 4; ++mi) {
            #pragma unroll
            for (int r = 0; r < 8; ++r) {
                const int mrow = m0 + wm + mi * 16 + l16 * 8 + r;
                float val = acc[mi][ni][r] + bv;
                const size_t oidx = ((size_t)e * MM + mrow) * (size_t)NN + ncol;
                if (GELU_OUT) {
                    val = 0.5f * val * (1.0f + erff(val * 0.70710678118654752f));
                    ((unsigned short*)Out)[oidx] = (unsigned short)f2bf1(val);
                } else {
                    ((float*)Out)[oidx] = val;
                }
            }
        }
    }
}

extern "C" void kernel_launch(void* const* d_in, const int* in_sizes, int n_in,
                              void* d_out, int out_size, void* d_ws, size_t ws_size,
                              hipStream_t stream) {
    const float* x  = (const float*)d_in[0];   // (E*C, H)
    const float* w1 = (const float*)d_in[1];   // (E, H, I)
    const float* b1 = (const float*)d_in[2];   // (E, I)
    const float* w2 = (const float*)d_in[3];   // (E, I, H)
    const float* b2 = (const float*)d_in[4];   // (E, H)

    // Intermediate h: (E, C, I) bf16 in workspace (64 MB)
    unsigned short* h = (unsigned short*)d_ws;

    dim3 blk(256, 1, 1);

    // GEMM1 + GELU: M=C, N=I, K=H ; A = x (fp32), out = h (bf16)
    dim3 g1(NI / 128, NC / 128, NE);
    moe_gemm_kernel<NC, NI, NH, false, true><<<g1, blk, 0, stream>>>(
        (const void*)x, w1, b1, (void*)h);

    // GEMM2 + bias: M=C, N=H, K=I ; A = h (bf16), out = d_out (fp32)
    dim3 g2(NH / 128, NC / 128, NE);
    moe_gemm_kernel<NC, NH, NI, true, false><<<g2, blk, 0, stream>>>(
        (const void*)h, w2, b2, d_out);
}
